// UpSample_17145509446491
// MI455X (gfx1250) — matrix operand order repaired and checked by reading, not compiled, not run on hardware
//
#include <hip/hip_runtime.h>
#include <hip/hip_bf16.h>
#include <stdint.h>

typedef __attribute__((ext_vector_type(16))) __bf16 v16bf;
typedef __attribute__((ext_vector_type(8)))  __bf16 v8bf;
typedef __attribute__((ext_vector_type(8)))  float  v8f;
typedef __attribute__((ext_vector_type(4)))  float  v4f;
typedef __attribute__((ext_vector_type(4)))  int    v4i;

constexpr int B_   = 4;
constexpr int H_   = 4;
constexpr int C_   = 128;
constexpr int NUP  = 8192;
constexpr int ND   = 2048;
constexpr int D_   = 32;                       // head depth
constexpr float QSCALE = 0.17677669529663689f; // 1/sqrt(32)
constexpr int WFRAG = 16384;                   // elems per fragment-ordered weight
constexpr int XPAD  = 136;                     // LDS row stride for X^T (keeps 16B align)

#if __has_builtin(__builtin_amdgcn_global_load_async_to_lds_b128) && \
    __has_builtin(__builtin_amdgcn_s_wait_asynccnt)
#define ASYNC_COPY 1
__device__ __forceinline__ void async_b128(const __bf16* g, __bf16* l) {
    auto gp = (__attribute__((address_space(1))) v4i*)(uintptr_t)g;
    auto lp = (__attribute__((address_space(3))) v4i*)(uint32_t)(uintptr_t)l;
    __builtin_amdgcn_global_load_async_to_lds_b128(gp, lp, 0, 0);
}
#endif

// ---- permute W (f32 row-major) into WMMA-A fragment order, f32 -> bf16 -------
// dst[w][mi*4+kc][lane][i] = W[o][c],  o = mi*16 + (lane&15),
// c = kc*32 + (lane>>4)*16 + i.  QSCALE folded into Wq.
__global__ __launch_bounds__(256) void convw(
    const float* __restrict__ Wq, const float* __restrict__ Wk,
    const float* __restrict__ Wv, const float* __restrict__ Wsk,
    __bf16* __restrict__ wf)
{
    int id   = blockIdx.x * 256 + threadIdx.x;   // 4 * 16384 total
    int w    = id >> 14;
    int rem  = id & (WFRAG - 1);
    int frag = rem >> 9;                          // mi*4 + kc
    int lane = (rem >> 4) & 31;
    int i    = rem & 15;
    int mi = frag >> 2, kc = frag & 3;
    int o  = mi * 16 + (lane & 15);
    int c  = kc * 32 + ((lane >> 4) << 4) + i;
    const float* W = (w == 0) ? Wq : (w == 1) ? Wk : (w == 2) ? Wv : Wsk;
    float val = W[o * C_ + c];
    if (w == 0) val *= QSCALE;
    wf[id] = (__bf16)val;
}

// ---- shared helper: stage f32 activation tile [128c][128n] as bf16 X^T[n][c] -
__device__ __forceinline__ void stage_xT(const float* __restrict__ src,
                                         int ldn, int n0, __bf16* XsT, int t)
{
    #pragma unroll
    for (int i = 0; i < 16; ++i) {
        int flat4 = t + 256 * i;                  // 4096 float4 = 128x128 f32
        int c = flat4 >> 5, n4 = (flat4 & 31) * 4;
        v4f x = *(const v4f*)&src[(size_t)c * ldn + n0 + n4];
        #pragma unroll
        for (int k = 0; k < 4; ++k)
            XsT[(n4 + k) * XPAD + c] = (__bf16)x[k];
    }
}

// ---- WMMA projection of pcd_up: q (bf16, pre-scaled) + skip (f32 -> d_out) ---
__global__ __launch_bounds__(256) void proj_x(
    const float* __restrict__ pcd, const __bf16* __restrict__ wfq,
    const __bf16* __restrict__ wfs, __bf16* __restrict__ qws,
    float* __restrict__ out)
{
    __shared__ __bf16 XsT[128 * XPAD];           // 34 KB
    const int blk = blockIdx.x;
    const int b   = blk / (NUP / 128);
    const int n0  = (blk % (NUP / 128)) * 128;
    const int t    = threadIdx.x;
    const int wave = t >> 5;
    const int lane = t & 31;
    const int half = lane >> 4;
    const int l16  = lane & 15;

    stage_xT(pcd + (size_t)b * C_ * NUP, NUP, n0, XsT, t);

    // resident A fragments: Wq row-tile `wave`, Wskip row-tile `wave`
    v16bf aq[4], as[4];
    #pragma unroll
    for (int kc = 0; kc < 4; ++kc) {
        aq[kc] = *(const v16bf*)&wfq[(wave * 4 + kc) * 512 + lane * 16];
        as[kc] = *(const v16bf*)&wfs[(wave * 4 + kc) * 512 + lane * 16];
    }
    __syncthreads();

    const int obase = wave * 16 + half * 8;      // C-layout row base (channel)
    const int hh = obase >> 5, dbase = obase & 31;

    for (int nj = 0; nj < 8; ++nj) {
        v8f accq = {}, accs = {};
        #pragma unroll
        for (int kc = 0; kc < 4; ++kc) {
            v16bf xb = *(const v16bf*)&XsT[(nj * 16 + l16) * XPAD + kc * 32 + half * 16];
            accq = __builtin_amdgcn_wmma_f32_16x16x32_bf16(
                       false, aq[kc], false, xb, (short)0, accq, false, false);
            accs = __builtin_amdgcn_wmma_f32_16x16x32_bf16(
                       false, as[kc], false, xb, (short)0, accs, false, true);
        }
        const int n = n0 + nj * 16 + l16;
        v8bf qv;
        #pragma unroll
        for (int r = 0; r < 8; ++r) qv[r] = (__bf16)accq[r];
        *(v8bf*)&qws[(((size_t)(b * H_ + hh)) * NUP + n) * D_ + dbase] = qv;
        #pragma unroll
        for (int r = 0; r < 8; ++r)
            out[((size_t)(b * C_ + obase + r)) * NUP + n] = accs[r];
    }
}

// ---- WMMA projection of points_select: k [bh][m][d], v transposed [bh][d][m] -
__global__ __launch_bounds__(256) void proj_kv(
    const float* __restrict__ ps, const __bf16* __restrict__ wfk,
    const __bf16* __restrict__ wfv, __bf16* __restrict__ kws,
    __bf16* __restrict__ vws)
{
    __shared__ __bf16 XsT[128 * XPAD];           // 34 KB
    const int blk = blockIdx.x;
    const int b   = blk / (ND / 128);
    const int m0  = (blk % (ND / 128)) * 128;
    const int t    = threadIdx.x;
    const int wave = t >> 5;
    const int lane = t & 31;
    const int half = lane >> 4;
    const int l16  = lane & 15;

    stage_xT(ps + (size_t)b * C_ * ND, ND, m0, XsT, t);

    v16bf ak[4], av[4];
    #pragma unroll
    for (int kc = 0; kc < 4; ++kc) {
        ak[kc] = *(const v16bf*)&wfk[(wave * 4 + kc) * 512 + lane * 16];
        av[kc] = *(const v16bf*)&wfv[(wave * 4 + kc) * 512 + lane * 16];
    }
    __syncthreads();

    const int obase = wave * 16 + half * 8;
    const int hh = obase >> 5, dbase = obase & 31;
    const size_t bh = (size_t)(b * H_ + hh);

    for (int nj = 0; nj < 8; ++nj) {
        v8f acck = {}, accv = {};
        #pragma unroll
        for (int kc = 0; kc < 4; ++kc) {
            v16bf xb = *(const v16bf*)&XsT[(nj * 16 + l16) * XPAD + kc * 32 + half * 16];
            acck = __builtin_amdgcn_wmma_f32_16x16x32_bf16(
                       false, ak[kc], false, xb, (short)0, acck, false, false);
            accv = __builtin_amdgcn_wmma_f32_16x16x32_bf16(
                       false, av[kc], false, xb, (short)0, accv, false, true);
        }
        const int m = m0 + nj * 16 + l16;
        v8bf kv;
        #pragma unroll
        for (int r = 0; r < 8; ++r) kv[r] = (__bf16)acck[r];
        *(v8bf*)&kws[(bh * ND + m) * D_ + dbase] = kv;
        #pragma unroll
        for (int r = 0; r < 8; ++r)
            vws[(bh * D_ + dbase + r) * ND + m] = (__bf16)accv[r];
    }
}

// ---------------- fused flash attention: out += softmax(qk) v -----------------
__global__ __launch_bounds__(256) void attn(
    const __bf16* __restrict__ qws, const __bf16* __restrict__ kws,
    const __bf16* __restrict__ vws, float* __restrict__ out)
{
    constexpr int TM = 128;   // query rows per block
    constexpr int TN = 128;   // keys per tile
    constexpr int NT = ND / TN;
    __shared__ __bf16 Ks[2 * TN * D_];   // double-buffered [key][d]  16 KB
    __shared__ __bf16 Vs[2 * D_ * TN];   // double-buffered [d][m]    16 KB
    __shared__ __bf16 Pb[8 * 16 * TN];   // per-wave P                32 KB

    const int blk  = blockIdx.x;
    const int nt   = blk % (NUP / TM);
    const int bh   = blk / (NUP / TM);
    const int b    = bh / H_;
    const int h    = bh % H_;
    const int t    = threadIdx.x;
    const int wave = t >> 5;
    const int lane = t & 31;
    const int half = lane >> 4;
    const int l16  = lane & 15;
    const int n0   = nt * TM + wave * 16;

    const __bf16* kbase = kws + ((size_t)bh) * ND * D_;
    const __bf16* vbase = vws + ((size_t)bh) * D_ * ND;

    const int k_key0 = (2 * t)     >> 2, k_ds0 = ((2 * t)     & 3) * 8;
    const int k_key1 = (2 * t + 1) >> 2, k_ds1 = ((2 * t + 1) & 3) * 8;
    const int v_d0   = (2 * t)     >> 4, v_mm0 = ((2 * t)     & 15) * 8;
    const int v_d1   = (2 * t + 1) >> 4, v_mm1 = ((2 * t + 1) & 15) * 8;

    auto stage = [&](int m0, int bufi) {
        __bf16* Kd = Ks + bufi * (TN * D_);
        __bf16* Vd = Vs + bufi * (D_ * TN);
#ifdef ASYNC_COPY
        async_b128(&kbase[(size_t)(m0 + k_key0) * D_ + k_ds0], &Kd[k_key0 * D_ + k_ds0]);
        async_b128(&kbase[(size_t)(m0 + k_key1) * D_ + k_ds1], &Kd[k_key1 * D_ + k_ds1]);
        async_b128(&vbase[(size_t)v_d0 * ND + m0 + v_mm0],     &Vd[v_d0 * TN + v_mm0]);
        async_b128(&vbase[(size_t)v_d1 * ND + m0 + v_mm1],     &Vd[v_d1 * TN + v_mm1]);
#else
        *(v8bf*)&Kd[k_key0 * D_ + k_ds0] = *(const v8bf*)&kbase[(size_t)(m0 + k_key0) * D_ + k_ds0];
        *(v8bf*)&Kd[k_key1 * D_ + k_ds1] = *(const v8bf*)&kbase[(size_t)(m0 + k_key1) * D_ + k_ds1];
        *(v8bf*)&Vd[v_d0 * TN + v_mm0]   = *(const v8bf*)&vbase[(size_t)v_d0 * ND + m0 + v_mm0];
        *(v8bf*)&Vd[v_d1 * TN + v_mm1]   = *(const v8bf*)&vbase[(size_t)v_d1 * ND + m0 + v_mm1];
#endif
    };

    // ---- Q fragment (A layout 16x32 bf16)
    const __bf16* qrow = qws + (((size_t)bh) * NUP + n0 + l16) * D_;
    v8bf qlo = *(const v8bf*)(qrow + half * 16);
    v8bf qhi = *(const v8bf*)(qrow + half * 16 + 8);
    v16bf qa;
    #pragma unroll
    for (int i = 0; i < 8; ++i) { qa[i] = qlo[i]; qa[8 + i] = qhi[i]; }

    v16bf ones;
    #pragma unroll
    for (int i = 0; i < 16; ++i) ones[i] = (__bf16)1.0f;

    v8f acc0 = {}, acc1 = {};
    float mrun[8], lrun[8];
    #pragma unroll
    for (int r = 0; r < 8; ++r) { mrun[r] = -INFINITY; lrun[r] = 0.f; }

    __bf16* pme = Pb + wave * 16 * TN;

    stage(0, 0);

    for (int it = 0; it < NT; ++it) {
        const int cur = it & 1;
        const __bf16* Kc = Ks + cur * (TN * D_);
        const __bf16* Vc = Vs + cur * (D_ * TN);

        if (it + 1 < NT) {
            stage((it + 1) * TN, (it + 1) & 1);
#ifdef ASYNC_COPY
            __builtin_amdgcn_s_wait_asynccnt(4);
#endif
        } else {
#ifdef ASYNC_COPY
            __builtin_amdgcn_s_wait_asynccnt(0);
#endif
        }
        __syncthreads();

        v8f s[8];
        {
            v16bf kb = *(const v16bf*)&Kc[(l16) * D_ + half * 16];
            v8f z = {};
            s[0] = __builtin_amdgcn_wmma_f32_16x16x32_bf16(
                       false, qa, false, kb, (short)0, z, false, false);
        }
        #pragma unroll
        for (int j = 1; j < 8; ++j) {
            v16bf kb = *(const v16bf*)&Kc[(j * 16 + l16) * D_ + half * 16];
            v8f z = {};
            s[j] = __builtin_amdgcn_wmma_f32_16x16x32_bf16(
                       false, qa, false, kb, (short)0, z, true, false);
        }

        float tmax[8];
        #pragma unroll
        for (int r = 0; r < 8; ++r) {
            float v = fmaxf(fmaxf(s[0][r], s[1][r]), fmaxf(s[2][r], s[3][r]));
            float w = fmaxf(fmaxf(s[4][r], s[5][r]), fmaxf(s[6][r], s[7][r]));
            tmax[r] = fmaxf(v, w);
        }
        #pragma unroll
        for (int msk = 1; msk < 16; msk <<= 1)
            #pragma unroll
            for (int r = 0; r < 8; ++r)
                tmax[r] = fmaxf(tmax[r], __shfl_xor(tmax[r], msk, 32));

        float alpha[8];
        #pragma unroll
        for (int r = 0; r < 8; ++r) {
            float mn = fmaxf(mrun[r], tmax[r]);
            alpha[r] = __expf(mrun[r] - mn);
            mrun[r]  = mn;
        }

        #pragma unroll
        for (int j = 0; j < 8; ++j)
            #pragma unroll
            for (int r = 0; r < 8; ++r) {
                float p = __expf(s[j][r] - mrun[r]);
                pme[(half * 8 + r) * TN + j * 16 + l16] = (__bf16)p;
            }
        #pragma unroll
        for (int r = 0; r < 8; ++r) { acc0[r] *= alpha[r]; acc1[r] *= alpha[r]; }
        __syncthreads();

        v8f rs = {};
        #pragma unroll
        for (int ks = 0; ks < 4; ++ks) {
            const __bf16* pr = pme + l16 * TN + ks * 32 + half * 16;
            v8bf plo = *(const v8bf*)(pr);
            v8bf phi = *(const v8bf*)(pr + 8);
            v16bf pa;
            #pragma unroll
            for (int i = 0; i < 8; ++i) { pa[i] = plo[i]; pa[8 + i] = phi[i]; }
            v16bf vb0 = *(const v16bf*)&Vc[(l16)      * TN + ks * 32 + half * 16];
            v16bf vb1 = *(const v16bf*)&Vc[(16 + l16) * TN + ks * 32 + half * 16];
            rs   = __builtin_amdgcn_wmma_f32_16x16x32_bf16(
                       false, pa, false, ones, (short)0, rs, false, false);
            acc0 = __builtin_amdgcn_wmma_f32_16x16x32_bf16(
                       false, pa, false, vb0, (short)0, acc0, true, false);
            acc1 = __builtin_amdgcn_wmma_f32_16x16x32_bf16(
                       false, pa, false, vb1, (short)0, acc1, true, false);
        }
        #pragma unroll
        for (int r = 0; r < 8; ++r)
            lrun[r] = lrun[r] * alpha[r] + rs[r];

        __syncthreads();
    }

    const int nbase = n0 + half * 8;
    float* op0 = out + ((size_t)(b * C_ + h * D_ + l16))      * NUP + nbase;
    float* op1 = out + ((size_t)(b * C_ + h * D_ + 16 + l16)) * NUP + nbase;
    #pragma unroll
    for (int r = 0; r < 8; ++r) {
        float inv = 1.0f / lrun[r];
        op0[r] += acc0[r] * inv;
        op1[r] += acc1[r] * inv;
    }
}

extern "C" void kernel_launch(void* const* d_in, const int* in_sizes, int n_in,
                              void* d_out, int out_size, void* d_ws, size_t ws_size,
                              hipStream_t stream) {
    const float* pcd = (const float*)d_in[0];
    const float* ps  = (const float*)d_in[1];
    const float* Wq  = (const float*)d_in[2];
    const float* Wk  = (const float*)d_in[3];
    const float* Wv  = (const float*)d_in[4];
    const float* Wsk = (const float*)d_in[5];
    float* out = (float*)d_out;

    char* ws = (char*)d_ws;
    size_t qbytes = (size_t)B_ * H_ * NUP * D_ * sizeof(__bf16);
    size_t kbytes = (size_t)B_ * H_ * ND  * D_ * sizeof(__bf16);
    __bf16* qws = (__bf16*)ws;
    __bf16* kws = (__bf16*)(ws + qbytes);
    __bf16* vws = (__bf16*)(ws + qbytes + kbytes);
    __bf16* wf  = (__bf16*)(ws + qbytes + 2 * kbytes);   // 4 * 16384 bf16
    __bf16* wfq = wf;
    __bf16* wfk = wf + WFRAG;
    __bf16* wfv = wf + 2 * WFRAG;
    __bf16* wfs = wf + 3 * WFRAG;

    convw  <<<(4 * WFRAG) / 256, 256, 0, stream>>>(Wq, Wk, Wv, Wsk, wf);
    proj_x <<<B_ * (NUP / 128), 256, 0, stream>>>(pcd, wfq, wfs, qws, out);
    proj_kv<<<B_ * (ND  / 128), 256, 0, stream>>>(ps,  wfk, wfv, kws, vws);
    attn   <<<B_ * H_ * (NUP / 128), 256, 0, stream>>>(qws, kws, vws, out);
}
